// LinAngularAttention_ksh_85676007621267
// MI455X (gfx1250) — matrix operand order repaired and checked by reading, not compiled, not run on hardware
//
#include <hip/hip_runtime.h>

typedef __attribute__((ext_vector_type(16))) __bf16 v16bf;
typedef __attribute__((ext_vector_type(8)))  float  v8f;

#define DIM     384
#define HEADS   8
#define HEAD_D  48
#define NTOK    4096
#define BATCH   8
#define MROWS   (BATCH * NTOK)          // 32768
#define KS      9

// ---------------------------------------------------------------------------
// helpers
// ---------------------------------------------------------------------------
__device__ inline v8f zero_v8f() {
    v8f z;
#pragma unroll
    for (int i = 0; i < 8; ++i) z[i] = 0.0f;
    return z;
}

__device__ inline v8f wmma_bf16(v16bf a, v16bf b, v8f c) {
    // (neg_a, A, neg_b, B, c_mod, C, reuse_a, reuse_b)
    return __builtin_amdgcn_wmma_f32_16x16x32_bf16(false, a, false, b, (short)0, c,
                                                   false, false);
}

// A fragment: 16x32 bf16, LDS tile row-major with row stride 32 elements.
// lane l: row m = base + (l & 15); lane[4] selects K-groups {0..7,16..23} vs {8..15,24..31}
__device__ inline v16bf load_frag_a32(const __bf16* s, int mOff, int lane) {
    const int m  = mOff + (lane & 15);
    const int hi = lane >> 4;
    union { uint4 u[2]; v16bf v; } r;
    r.u[0] = *(const uint4*)(s + m * 32 + hi * 8);        // K = hi*8 .. +7
    r.u[1] = *(const uint4*)(s + m * 32 + 16 + hi * 8);   // K = 16 + hi*8 .. +7
    return r.v;
}

// B fragment: 32x16 bf16, LDS tile stored transposed: rows = N, row stride 32 (=K).
// lane l: col n = base + (l & 15); lane[4] selects K-groups {0..7,16..23} vs {8..15,24..31}
__device__ inline v16bf load_frag_b32(const __bf16* s, int nOff, int lane) {
    const int n  = nOff + (lane & 15);
    const int hi = lane >> 4;
    union { uint4 u[2]; v16bf v; } r;
    r.u[0] = *(const uint4*)(s + n * 32 + hi * 8);
    r.u[1] = *(const uint4*)(s + n * 32 + 16 + hi * 8);
    return r.v;
}

// ---------------------------------------------------------------------------
// fp32 -> bf16 convert
// ---------------------------------------------------------------------------
__global__ void cvt_f32_bf16(const float* __restrict__ src, __bf16* __restrict__ dst, int n) {
    int i = blockIdx.x * blockDim.x + threadIdx.x;
    if (i < n) dst[i] = (__bf16)src[i];
}

// ---------------------------------------------------------------------------
// NT GEMM: C[m,n] = sum_k A[m,k] * W[n,k]      (both operands bf16, K-major)
// block tile 128(M) x 128(N), 256 threads = 8 waves (4M x 2N), wave = 32x64 =
// 2x4 fragments. Double-buffered LDS (one barrier per 32-deep K step) with
// global->register prefetch of the next tile while WMMAs issue.
//
// MODE 0 (fused qk+v projection): W has 768 rows (w_qk ++ w_v); block-uniform
//        split: nBase<384 -> outS = sign(acc) bf16 ; else outF = acc fp32.
// MODE 1 (final proj): outF = acc + bias[n].
// ---------------------------------------------------------------------------
template <int MODE>
__global__ __launch_bounds__(256)
void gemm_nt(const __bf16* __restrict__ A, const __bf16* __restrict__ W,
             int K, int ldc,
             float* __restrict__ outF, __bf16* __restrict__ outS,
             const float* __restrict__ bias) {
    const int mBase = blockIdx.x * 128;
    const int nBase = blockIdx.y * 128;
    const int tid   = threadIdx.x;
    const int wave  = tid >> 5;
    const int lane  = tid & 31;
    const int mw    = (wave >> 1) * 32;   // 0,32,64,96
    const int nw    = (wave & 1) * 64;    // 0,64

    __shared__ __align__(16) __bf16 As[2][128 * 32];
    __shared__ __align__(16) __bf16 Bs[2][128 * 32];

    v8f acc[2][4];
#pragma unroll
    for (int mi = 0; mi < 2; ++mi)
#pragma unroll
        for (int ni = 0; ni < 4; ++ni) acc[mi][ni] = zero_v8f();

    const int row = tid >> 1;       // 0..127
    const int seg = tid & 1;        // 16 bf16 per segment
    const __bf16* aRow = A + (size_t)(mBase + row) * K + seg * 16;
    const __bf16* bRow = W + (size_t)(nBase + row) * K + seg * 16;

    // prologue: stage K-step 0 into buffer 0
    {
        const uint4* ga = (const uint4*)aRow;
        const uint4* gb = (const uint4*)bRow;
        uint4* sa = (uint4*)(As[0] + row * 32 + seg * 16);
        uint4* sb = (uint4*)(Bs[0] + row * 32 + seg * 16);
        sa[0] = ga[0];  sa[1] = ga[1];
        sb[0] = gb[0];  sb[1] = gb[1];
    }
    __syncthreads();

    int buf = 0;
    for (int k0 = 0; k0 < K; k0 += 32) {
        const int kn = k0 + 32;
        uint4 ra0, ra1, rb0, rb1;
        if (kn < K) {
            const uint4* ga = (const uint4*)(aRow + kn);
            const uint4* gb = (const uint4*)(bRow + kn);
            ra0 = ga[0];  ra1 = ga[1];
            rb0 = gb[0];  rb1 = gb[1];
            if (kn + 32 < K) {
                __builtin_prefetch(aRow + kn + 32, 0, 1);   // global_prefetch_b8
                __builtin_prefetch(bRow + kn + 32, 0, 1);
            }
        }

        v16bf afr[2], bfr[4];
#pragma unroll
        for (int mi = 0; mi < 2; ++mi) afr[mi] = load_frag_a32(As[buf], mw + mi * 16, lane);
#pragma unroll
        for (int ni = 0; ni < 4; ++ni) bfr[ni] = load_frag_b32(Bs[buf], nw + ni * 16, lane);
#pragma unroll
        for (int mi = 0; mi < 2; ++mi)
#pragma unroll
            for (int ni = 0; ni < 4; ++ni)
                acc[mi][ni] = wmma_bf16(afr[mi], bfr[ni], acc[mi][ni]);

        if (kn < K) {
            uint4* sa = (uint4*)(As[buf ^ 1] + row * 32 + seg * 16);
            uint4* sb = (uint4*)(Bs[buf ^ 1] + row * 32 + seg * 16);
            sa[0] = ra0;  sa[1] = ra1;
            sb[0] = rb0;  sb[1] = rb1;
            buf ^= 1;
            __syncthreads();
        }
    }

    // epilogue: C/D layout -> lane: n = l&15 ; VGPR r: m = r + 8*(l>>4)
    const bool isQK = (MODE == 0) && (nBase < DIM);   // block-uniform
#pragma unroll
    for (int mi = 0; mi < 2; ++mi) {
#pragma unroll
        for (int ni = 0; ni < 4; ++ni) {
            const int n = nBase + nw + ni * 16 + (lane & 15);
#pragma unroll
            for (int r = 0; r < 8; ++r) {
                const int m = mBase + mw + mi * 16 + r + 8 * (lane >> 4);
                const float val = acc[mi][ni][r];
                if (MODE == 0) {
                    if (isQK) {
                        outS[(size_t)m * ldc + n] =
                            (__bf16)((val > 0.0f) ? 1.0f : ((val < 0.0f) ? -1.0f : 0.0f));
                    } else {
                        outF[(size_t)m * ldc + (n - DIM)] = val;
                    }
                } else {
                    outF[(size_t)m * ldc + n] = val + bias[n];
                }
            }
        }
    }
}

// ---------------------------------------------------------------------------
// attn[b,h,d,e] = sum_n qk[b,n,h,d] * v[b,n,h,e]     (48x48, K = 4096)
// 1 block per (b,h); 4 waves, each covers 1024 tokens (32 iters of 32),
// 3x3 fragment grid; cross-wave reduction via LDS float atomics.
// ---------------------------------------------------------------------------
__global__ __launch_bounds__(128)
void attn_kernel(const __bf16* __restrict__ qk_q, const float* __restrict__ v_f,
                 float* __restrict__ attn) {
    const int b = blockIdx.x >> 3;
    const int h = blockIdx.x & 7;
    const int wave = threadIdx.x >> 5;
    const int lane = threadIdx.x & 31;

    __shared__ __align__(16) __bf16 qkT[4][48 * 32];   // [d][tok]
    __shared__ __align__(16) __bf16 vT[4][48 * 32];    // [e][tok]
    __shared__ float attn_s[48 * 48];

    for (int i = threadIdx.x; i < 48 * 48; i += 128) attn_s[i] = 0.0f;
    __syncthreads();

    v8f acc[3][3];
#pragma unroll
    for (int di = 0; di < 3; ++di)
#pragma unroll
        for (int ei = 0; ei < 3; ++ei) acc[di][ei] = zero_v8f();

    const int tok0 = wave * 1024;
    __bf16* qT  = qkT[wave];
    __bf16* vTt = vT[wave];

    for (int it = 0; it < 32; ++it) {
        const int tokBase = tok0 + it * 32;
        const size_t rowOff = (size_t)(b * NTOK + tokBase + lane) * DIM + h * HEAD_D;
        const __bf16* qg = qk_q + rowOff;
        const float*  vg = v_f + rowOff;
        __builtin_prefetch(qg + 32 * DIM, 0, 1);
        __builtin_prefetch(vg + 32 * DIM, 0, 1);
        // stage transposed: lane <-> token, loop over 48 channels
#pragma unroll 4
        for (int c = 0; c < 48; ++c) {
            qT[c * 32 + lane]  = qg[c];
            vTt[c * 32 + lane] = (__bf16)vg[c];
        }
        __syncthreads();

        v16bf afr[3], bfr[3];
#pragma unroll
        for (int di = 0; di < 3; ++di) afr[di] = load_frag_a32(qT, di * 16, lane);
#pragma unroll
        for (int ei = 0; ei < 3; ++ei) bfr[ei] = load_frag_b32(vTt, ei * 16, lane);
#pragma unroll
        for (int di = 0; di < 3; ++di)
#pragma unroll
            for (int ei = 0; ei < 3; ++ei)
                acc[di][ei] = wmma_bf16(afr[di], bfr[ei], acc[di][ei]);
        __syncthreads();
    }

    // reduce 4 wave-partials into LDS
#pragma unroll
    for (int di = 0; di < 3; ++di) {
#pragma unroll
        for (int ei = 0; ei < 3; ++ei) {
            const int e = ei * 16 + (lane & 15);
#pragma unroll
            for (int r = 0; r < 8; ++r) {
                const int d = di * 16 + r + 8 * (lane >> 4);
                atomicAdd(&attn_s[d * 48 + e], acc[di][ei][r]);
            }
        }
    }
    __syncthreads();

    float* ag = attn + (size_t)(b * HEADS + h) * 48 * 48;
    for (int i = threadIdx.x; i < 48 * 48; i += 128) ag[i] = attn_s[i];
}

// ---------------------------------------------------------------------------
// out_pre[b,n,h,e] = normalize_e( 0.5*v + (1/pi) * sum_d qk[n,d]*attn[d,e] )
//                    + sum_j w[h,j]*v[n+j-4,e]          (stored bf16)
// 1 block per (b,h,64-token chunk); 64 threads, one token per thread.
// ---------------------------------------------------------------------------
__global__ __launch_bounds__(64)
void combine_kernel(const __bf16* __restrict__ qk_q, const float* __restrict__ v_f,
                    const float* __restrict__ attn, const float* __restrict__ dconv_w,
                    __bf16* __restrict__ out_pre) {
    const int blk   = blockIdx.x;
    const int chunk = blk & 63;
    const int h     = (blk >> 6) & 7;
    const int b     = blk >> 9;
    const int tokBase = chunk * 64;

    __shared__ float  attn_s[48 * 48];
    __shared__ __bf16 qk_s[64 * 48];
    __shared__ float  v_s[72 * 48];     // 4-token halo each side
    __shared__ float  w_s[KS];

    const float* ag = attn + (size_t)(b * HEADS + h) * 2304;
    for (int i = threadIdx.x; i < 2304; i += 64) attn_s[i] = ag[i];
    for (int i = threadIdx.x; i < 64 * 48; i += 64) {
        const int t = i / 48, c = i % 48;
        qk_s[i] = qk_q[(size_t)(b * NTOK + tokBase + t) * DIM + h * HEAD_D + c];
    }
    for (int i = threadIdx.x; i < 72 * 48; i += 64) {
        const int t = i / 48, c = i % 48;
        const int tok = tokBase + t - 4;
        v_s[i] = (tok >= 0 && tok < NTOK)
                     ? v_f[(size_t)(b * NTOK + tok) * DIM + h * HEAD_D + c]
                     : 0.0f;
    }
    if (threadIdx.x < KS) w_s[threadIdx.x] = dconv_w[h * KS + threadIdx.x];
    __syncthreads();

    const int t = threadIdx.x;          // token within chunk
    float o[48];
#pragma unroll
    for (int e = 0; e < 48; ++e) o[e] = 0.5f * v_s[(t + 4) * 48 + e];

    const float invpi = 0.318309886183790671537767526745f;
    for (int d = 0; d < 48; ++d) {
        const float q = (float)qk_s[t * 48 + d] * invpi;
#pragma unroll 8
        for (int e = 0; e < 48; ++e) o[e] += q * attn_s[d * 48 + e];
    }

    float ss = 0.0f;
#pragma unroll
    for (int e = 0; e < 48; ++e) ss += o[e] * o[e];
    const float rn = rsqrtf(ss);

    __bf16* og = out_pre + (size_t)(b * NTOK + tokBase + t) * DIM + h * HEAD_D;
#pragma unroll 4
    for (int e = 0; e < 48; ++e) {
        float conv = 0.0f;
#pragma unroll
        for (int j = 0; j < KS; ++j) conv += w_s[j] * v_s[(t + j) * 48 + e];
        og[e] = (__bf16)(o[e] * rn + conv);
    }
}

// ---------------------------------------------------------------------------
// launch
// ---------------------------------------------------------------------------
extern "C" void kernel_launch(void* const* d_in, const int* in_sizes, int n_in,
                              void* d_out, int out_size, void* d_ws, size_t ws_size,
                              hipStream_t stream) {
    const float* x       = (const float*)d_in[0];
    const float* w_qk    = (const float*)d_in[3];
    const float* w_v     = (const float*)d_in[4];
    const float* w_proj  = (const float*)d_in[5];
    const float* b_proj  = (const float*)d_in[6];
    const float* dconv_w = (const float*)d_in[7];

    // workspace layout (256B aligned)
    char* ws = (char*)d_ws;
    size_t off = 0;
    auto alloc = [&](size_t bytes) -> char* {
        char* p = ws + off;
        off = (off + bytes + 255) & ~(size_t)255;
        return p;
    };
    const size_t nX = (size_t)MROWS * DIM;        // 12,582,912
    const size_t nW = (size_t)DIM * DIM;          // 147,456

    __bf16* x_bf    = (__bf16*)alloc(nX * 2);
    __bf16* wcat_bf = (__bf16*)alloc(2 * nW * 2);   // rows 0..383 = w_qk, 384..767 = w_v
    __bf16* wp_bf   = (__bf16*)alloc(nW * 2);
    __bf16* qk_q    = (__bf16*)alloc(nX * 2);
    float*  v_f     = (float*)alloc(nX * 4);
    float*  attn    = (float*)alloc((size_t)BATCH * HEADS * 48 * 48 * 4);
    __bf16* out_pre = (__bf16*)alloc(nX * 2);

    // 1. converts
    cvt_f32_bf16<<<(int)((nX + 255) / 256), 256, 0, stream>>>(x, x_bf, (int)nX);
    cvt_f32_bf16<<<(int)((nW + 255) / 256), 256, 0, stream>>>(w_qk, wcat_bf, (int)nW);
    cvt_f32_bf16<<<(int)((nW + 255) / 256), 256, 0, stream>>>(w_v, wcat_bf + nW, (int)nW);
    cvt_f32_bf16<<<(int)((nW + 255) / 256), 256, 0, stream>>>(w_proj, wp_bf, (int)nW);

    // 2. fused qk+v projection (WMMA): one pass over x_bf, N = 768
    dim3 gProj(MROWS / 128, (2 * DIM) / 128);
    gemm_nt<0><<<gProj, 256, 0, stream>>>(x_bf, wcat_bf, DIM, DIM,
                                          v_f, qk_q, nullptr);

    // 3. attn = qk^T v per (b,h)  (WMMA)
    attn_kernel<<<BATCH * HEADS, 128, 0, stream>>>(qk_q, v_f, attn);

    // 4. combine: 0.5v + (1/pi) qk*attn, normalize, + depthwise conv
    combine_kernel<<<BATCH * HEADS * (NTOK / 64), 64, 0, stream>>>(
        qk_q, v_f, attn, dconv_w, out_pre);

    // 5. final projection + bias (WMMA) -> fp32 output
    dim3 gOut(MROWS / 128, DIM / 128);
    gemm_nt<1><<<gOut, 256, 0, stream>>>(out_pre, wp_bf, DIM, DIM,
                                         (float*)d_out, nullptr, b_proj);
}
